// ConvCaps_80341658239479
// MI455X (gfx1250) — matrix operand order repaired and checked by reading, not compiled, not run on hardware
//
#include <hip/hip_runtime.h>
#include <hip/hip_bf16.h>
#include <math.h>

// ---------------------------------------------------------------------------
// Shapes (fixed by the reference):
//   b=8, H=W=64, B=C=32, K=3, PSIZE=16, stride=2 -> oh=ow=32, hw=1024, kk=9
//   M = 8*32*32 = 8192 rows
//   GEMM1: M x 4608 @ 4608 x 288   (P * W1^T)
//   GEMM2: M x 4896 @ 4896 x 544   ([P | a] * W2^T)
// ---------------------------------------------------------------------------

typedef __attribute__((ext_vector_type(16))) __bf16 v16bf;
typedef __attribute__((ext_vector_type(8)))  float  v8f;

union Frag32B { v16bf v; uint4 q[2]; };

__device__ __forceinline__ unsigned short f2bf(float f) {
    unsigned int u = __float_as_uint(f);
    unsigned int r = u + 0x7FFFu + ((u >> 16) & 1u);   // round-to-nearest-even
    return (unsigned short)(r >> 16);
}

__device__ __forceinline__ v8f wmma_bf16(const Frag32B& a, const Frag32B& b, v8f c) {
    return __builtin_amdgcn_wmma_f32_16x16x32_bf16(false, a.v, false, b.v,
                                                   (short)0, c, false, false);
}

// ---------------------------------------------------------------------------
// Kernel: f32 weights (N x K row-major) -> bf16 TRANSPOSED (K x N row-major).
// B-matrix WMMA layout wants lane=n with packed values running over k, so
// K-major rows make both staging and fragment loads contiguous.
// ---------------------------------------------------------------------------
__global__ void cvt_transpose_bf16_kernel(const float* __restrict__ src, // N x K
                                          unsigned short* __restrict__ dst, // K x N
                                          int N, int K) {
    int i = blockIdx.x * 256 + threadIdx.x;   // index over dst, n fastest
    if (i >= N * K) return;
    int k = i / N, n = i - k * N;
    dst[i] = f2bf(src[(size_t)n * K + k]);
}

// ---------------------------------------------------------------------------
// Kernel: fold softmax(weights over C), softmax(gamma over B) and gamma into
// one 32x32 effective mixing matrix:
//   weff[c][d] = (w_sm[c][d] + w_sm[c][32]*gsm[d]) * gamma[d]
// ---------------------------------------------------------------------------
__global__ void weff_kernel(const float* __restrict__ weights,  // (32,33) row-major
                            const float* __restrict__ gamma,    // (32)
                            float* __restrict__ weff) {
    if (threadIdx.x != 0 || blockIdx.x != 0) return;
    float gsm[32];
    float gmax = -1e30f;
    for (int d = 0; d < 32; ++d) gmax = fmaxf(gmax, gamma[d]);
    float gs = 0.f;
    for (int d = 0; d < 32; ++d) { gsm[d] = expf(gamma[d] - gmax); gs += gsm[d]; }
    for (int d = 0; d < 32; ++d) gsm[d] /= gs;

    float wsm[32][33];
    for (int d = 0; d < 33; ++d) {
        float mx = -1e30f;
        for (int c = 0; c < 32; ++c) mx = fmaxf(mx, weights[c * 33 + d]);
        float s = 0.f;
        for (int c = 0; c < 32; ++c) { wsm[c][d] = expf(weights[c * 33 + d] - mx); s += wsm[c][d]; }
        for (int c = 0; c < 32; ++c) wsm[c][d] /= s;
    }
    for (int c = 0; c < 32; ++c)
        for (int d = 0; d < 32; ++d)
            weff[c * 32 + d] = (wsm[c][d] + wsm[c][32] * gsm[d]) * gamma[d];
}

// ---------------------------------------------------------------------------
// Kernel: weighted LayerNorm statistics per (batch, capsule).
// Patch-extracted mean/var equal a weighted sum over the 64x64 image with
// weight cnt(i+1)*cnt(j+1), cnt(r) = #patch windows covering padded row r.
// N = 96*96*16 = 147456 (zero padding contributes zeros but counts toward N).
// ---------------------------------------------------------------------------
__global__ __launch_bounds__(256)
void stats_kernel(const float* __restrict__ x, float* __restrict__ muB,
                  float* __restrict__ rsB) {
    __shared__ float cw[66];
    __shared__ float rs[256], rss[256];
    int tid = threadIdx.x;
    if (tid < 66) {
        int r = tid;
        int lo = r - 2; if (lo < 0) lo = 0;
        int hi = r;     if (hi > 62) hi = 62;
        int cnt = (hi < lo) ? 0 : (hi / 2 - (lo + 1) / 2 + 1);
        cw[r] = (float)cnt;
    }
    __syncthreads();
    int b = blockIdx.x >> 5, cap = blockIdx.x & 31;
    const float* base = x + (((size_t)b * 544 + cap * 16) * 4096);
    float s = 0.f, ss = 0.f;
    for (int pix = tid; pix < 4096; pix += 256) {
        int h = pix >> 6, w = pix & 63;
        float wgt = cw[h + 1] * cw[w + 1];
        #pragma unroll
        for (int p = 0; p < 16; ++p) {
            float v = base[(size_t)p * 4096 + pix];
            s  += wgt * v;
            ss += wgt * v * v;
        }
    }
    rs[tid] = s; rss[tid] = ss;
    __syncthreads();
    for (int st = 128; st > 0; st >>= 1) {
        if (tid < st) { rs[tid] += rs[tid + st]; rss[tid] += rss[tid + st]; }
        __syncthreads();
    }
    if (tid == 0) {
        const float N = 147456.0f;
        float S = rs[0], SS = rss[0];
        float mu  = S / N;
        float var = (SS - S * S / N) / (N - 1.0f);
        muB[blockIdx.x] = mu;
        rsB[blockIdx.x] = rsqrtf(var + 1e-8f);
    }
}

// ---------------------------------------------------------------------------
// Kernel: build P (8192 x 4608, bf16). One block per output spatial position.
//   P[m, kk*512 + c*16 + p] = sum_d weff[c][d]*a[kk][d]*(x[kk][d*16+p]-mu[d])*rstd[d]
// ---------------------------------------------------------------------------
__global__ __launch_bounds__(256)
void build_p_kernel(const float* __restrict__ x, const float* __restrict__ weff,
                    const float* __restrict__ muB, const float* __restrict__ rsB,
                    unsigned short* __restrict__ P) {
    __shared__ float xp[9 * 544];
    __shared__ float wf[1024];
    __shared__ float mu[32], rsd[32];
    int tid = threadIdx.x;
    int m = blockIdx.x;
    int b = m >> 10, h = (m >> 5) & 31, w = m & 31;

    for (int i = tid; i < 1024; i += 256) wf[i] = weff[i];
    if (tid < 32) { mu[tid] = muB[b * 32 + tid]; rsd[tid] = rsB[b * 32 + tid]; }
    for (int i = tid; i < 9 * 544; i += 256) {
        int kk = i / 544, ch = i - kk * 544;
        int ki = kk / 3, kj = kk - ki * 3;
        int ih = 2 * h + ki - 1, iw = 2 * w + kj - 1;
        float v = 0.0f;
        if (ih >= 0 && ih < 64 && iw >= 0 && iw < 64)
            v = x[(((size_t)b * 544 + ch) * 64 + ih) * 64 + iw];
        xp[i] = v;
    }
    __syncthreads();

    unsigned short* dst = P + (size_t)m * 4608;
    for (int o = tid; o < 4608; o += 256) {
        int kk = o >> 9, rem = o & 511, c = rem >> 4, p = rem & 15;
        const float* xk = xp + kk * 544;
        float acc = 0.f;
        #pragma unroll 8
        for (int d = 0; d < 32; ++d) {
            float pv = (xk[d * 16 + p] - mu[d]) * rsd[d];
            acc += wf[c * 32 + d] * xk[512 + d] * pv;
        }
        dst[o] = f2bf(acc);
    }
}

// ---------------------------------------------------------------------------
// One LDS-staged K-tile of the WMMA GEMM. KW = 64 (main) or 32 (tail).
// Straight-line staging: every thread owns a compile-time-fixed set of
// 16-byte segments; all global loads issue as one clause BEFORE the barrier
// (overlapping prior tile's WMMA compute), LDS stores after it.
//   A tile: 128 rows x KW cols (row-major, from A, row stride lda)
//   B tile: KW k-rows x 32 n-cols (row-major, from transposed weights Wt)
// Fragment layouts per CDNA5 ISA (bf16, wave32):
//   A (16x32): lane L -> m=L&15, kbase=(L>=16)*8; 16B at k=kbase, 16B at 16+kbase
//   B (32x16): lane L -> n=L&15, same k order as A (kbase..+7, 16+kbase..+7)
// ---------------------------------------------------------------------------
template <int KW>
__device__ __forceinline__
void gemm_tile(const unsigned short* __restrict__ Abase, int lda, int kA,
               const unsigned short* __restrict__ Wt, int ldb, int n0, int kB,
               unsigned short* Alds, unsigned short* Blds,
               int tid, int lane, int waveM, int waveN,
               v8f& acc0, v8f& acc1) {
    constexpr int SEGROW = KW / 8;        // 16B segments per A row (8 or 4)
    constexpr int NA     = KW / 16;       // A segments per thread (4 or 2)
    constexpr int RSTEP  = 256 / SEGROW;  // row stride between a thread's segs
    constexpr int BSEGS  = KW * 4;        // B segments total (256 or 128)

    const int segA  = tid % SEGROW;
    const int rowA0 = tid / SEGROW;
    const int segB  = tid & 3;
    const int rowB  = tid >> 2;

    // ---- global loads (issue as a clause; overlap prior compute) ----
    uint4 areg[NA];
    #pragma unroll
    for (int i = 0; i < NA; ++i)
        areg[i] = *(const uint4*)(Abase + (size_t)(rowA0 + i * RSTEP) * lda
                                        + kA + segA * 8);
    uint4 breg = {};
    const bool hasB = (BSEGS == 256) || (tid < BSEGS);
    if (hasB)
        breg = *(const uint4*)(Wt + (size_t)(kB + rowB) * ldb + n0 + segB * 8);

    // prefetch next A tile into L2 (global_prefetch_b8)
    __builtin_prefetch((const void*)(Abase + (size_t)rowA0 * lda + kA + KW), 0, 1);

    __syncthreads();   // previous tile's compute done -> safe to overwrite LDS

    #pragma unroll
    for (int i = 0; i < NA; ++i)
        *(uint4*)(Alds + (rowA0 + i * RSTEP) * KW + segA * 8) = areg[i];
    if (hasB)
        *(uint4*)(Blds + rowB * 32 + segB * 8) = breg;

    __syncthreads();

    // ---- WMMA compute ----
    const int mf = lane & 15;
    const int kbase = (lane >> 4) << 3;
    #pragma unroll
    for (int kh = 0; kh < KW / 32; ++kh) {
        Frag32B bfr, a0, a1;
        const unsigned short* bp = Blds + (kh * 32 + kbase) * 32 + waveN * 16 + mf;
        unsigned int* bw = (unsigned int*)bfr.q;
        #pragma unroll
        for (int j = 0; j < 8; ++j) {
            const int kq = (j < 4) ? 2 * j : 16 + 2 * (j - 4);
            bw[j] = (unsigned int)bp[kq * 32] | ((unsigned int)bp[(kq + 1) * 32] << 16);
        }
        const unsigned short* ap = Alds + (waveM * 32 + mf) * KW + kh * 32;
        a0.q[0] = *(const uint4*)(ap + kbase);
        a0.q[1] = *(const uint4*)(ap + 16 + kbase);
        const unsigned short* ap1 = ap + 16 * KW;
        a1.q[0] = *(const uint4*)(ap1 + kbase);
        a1.q[1] = *(const uint4*)(ap1 + 16 + kbase);
        acc0 = wmma_bf16(a0, bfr, acc0);
        acc1 = wmma_bf16(a1, bfr, acc1);
    }
}

// ---------------------------------------------------------------------------
// WMMA bf16 GEMM with fused BN epilogue.
//   D[m,n] = sum_k A[m,k] * Wt[k,n]    (A split into segments A0 | A1)
// Block: 256 threads (8 waves), tile 128(M) x 32(N), K-step 64 (+32 tail).
// Each wave owns 32x16 of the output (two stacked 16x16 WMMA tiles).
// mode 0: sigmoid(bn(.)) -> bf16 row-major (ld = Nld)    [GEMM1 -> a]
// mode 1: relu(bn(.))    -> f32 NCHW transposed store    [GEMM2 -> y]
// ---------------------------------------------------------------------------
__global__ __launch_bounds__(256)
void gemm_bf16_kernel(const unsigned short* __restrict__ A0, int K0,
                      const unsigned short* __restrict__ A1, int K1,
                      const unsigned short* __restrict__ Wt, int N,
                      const float* __restrict__ bnS, const float* __restrict__ bnB,
                      const float* __restrict__ bnM, const float* __restrict__ bnV,
                      unsigned short* __restrict__ outBf, float* __restrict__ outF,
                      int Nld, int mode) {
    __shared__ unsigned short Alds[128 * 64];   // 16 KB
    __shared__ unsigned short Blds[64 * 32];    //  4 KB

    const int tid   = threadIdx.x;
    const int lane  = tid & 31;
    const int wave  = tid >> 5;
    const int waveM = wave >> 1;      // 0..3
    const int waveN = wave & 1;       // 0..1
    const int blockM = blockIdx.x * 128;
    const int blockN = blockIdx.y * 32;

    v8f acc0 = {};
    v8f acc1 = {};

    const unsigned short* Abase0 = A0 + (size_t)blockM * K0;
    for (int k = 0; k < K0; k += 64)
        gemm_tile<64>(Abase0, K0, k, Wt, N, blockN, k,
                      Alds, Blds, tid, lane, waveM, waveN, acc0, acc1);

    if (A1 != nullptr) {
        const unsigned short* Abase1 = A1 + (size_t)blockM * K1;
        int k = 0;
        for (; k + 64 <= K1; k += 64)
            gemm_tile<64>(Abase1, K1, k, Wt, N, blockN, K0 + k,
                          Alds, Blds, tid, lane, waveM, waveN, acc0, acc1);
        if (k < K1)   // all K dims are multiples of 32 -> exactly one 32 tail
            gemm_tile<32>(Abase1, K1, k, Wt, N, blockN, K0 + k,
                          Alds, Blds, tid, lane, waveM, waveN, acc0, acc1);
    }

    // Epilogue. C/D layout: VGPR r, lanes<16 -> M=r, lanes>=16 -> M=8+r; N=lane&15
    const int n    = blockN + waveN * 16 + (lane & 15);   // constant per lane
    const int mhi  = (lane >> 4) * 8;
    const float sc = bnS[n] * rsqrtf(bnV[n] + 1e-5f);
    const float bm = bnM[n];
    const float bb2 = bnB[n];
    #pragma unroll
    for (int r = 0; r < 8; ++r) {
        #pragma unroll
        for (int t = 0; t < 2; ++t) {
            float v = (t == 0) ? acc0[r] : acc1[r];
            int m = blockM + waveM * 32 + t * 16 + r + mhi;
            float tval = (v - bm) * sc + bb2;
            if (mode == 0) {
                float s = 1.0f / (1.0f + expf(-tval));
                outBf[(size_t)m * Nld + n] = f2bf(s);
            } else {
                float rel = fmaxf(tval, 0.0f);
                int bi = m >> 10, hh = (m >> 5) & 31, ww = m & 31;
                outF[(((size_t)bi * 544 + n) * 32 + hh) * 32 + ww] = rel;
            }
        }
    }
}

// ---------------------------------------------------------------------------
// Host launcher
// ---------------------------------------------------------------------------
extern "C" void kernel_launch(void* const* d_in, const int* in_sizes, int n_in,
                              void* d_out, int out_size, void* d_ws, size_t ws_size,
                              hipStream_t stream) {
    (void)in_sizes; (void)n_in; (void)out_size; (void)ws_size;

    const float* x        = (const float*)d_in[0];   // (8, 544, 64, 64)
    const float* weights  = (const float*)d_in[1];   // (1,1,32,33)
    const float* gamma    = (const float*)d_in[2];   // (1,1,1,32,1)
    const float* conv1_w  = (const float*)d_in[3];   // (288, 4608)
    const float* bn1S     = (const float*)d_in[4];
    const float* bn1B     = (const float*)d_in[5];
    const float* bn1M     = (const float*)d_in[6];
    const float* bn1V     = (const float*)d_in[7];
    const float* conv2_w  = (const float*)d_in[8];   // (544, 4896)
    const float* bn2S     = (const float*)d_in[9];
    const float* bn2B     = (const float*)d_in[10];
    const float* bn2M     = (const float*)d_in[11];
    const float* bn2V     = (const float*)d_in[12];
    float* out = (float*)d_out;                      // (8, 544, 32, 32)

    const int M = 8192;
    const int K0 = 4608, N1 = 288;
    const int K1 = 288,  N2 = 544, KT2 = 4896;
    const int nW1 = N1 * K0;            // 1,327,104
    const int nW2 = N2 * KT2;           // 2,663,424

    // Workspace layout (256-byte aligned)
    char* ws = (char*)d_ws;
    size_t off = 0;
    auto take = [&](size_t bytes) { char* p = ws + off; off = (off + bytes + 255) & ~(size_t)255; return p; };
    unsigned short* W1t  = (unsigned short*)take((size_t)nW1 * 2);      // 4608 x 288
    unsigned short* W2t  = (unsigned short*)take((size_t)nW2 * 2);      // 4896 x 544
    unsigned short* P    = (unsigned short*)take((size_t)M * K0 * 2);   // 75.5 MB
    unsigned short* Abuf = (unsigned short*)take((size_t)M * K1 * 2);   // 4.7 MB
    float*          muB  = (float*)take(256 * 4);
    float*          rsB  = (float*)take(256 * 4);
    float*          weff = (float*)take(1024 * 4);

    // 1) Weights -> bf16, transposed to K-major (WMMA B-operand layout)
    cvt_transpose_bf16_kernel<<<(nW1 + 255) / 256, 256, 0, stream>>>(conv1_w, W1t, N1, K0);
    cvt_transpose_bf16_kernel<<<(nW2 + 255) / 256, 256, 0, stream>>>(conv2_w, W2t, N2, KT2);

    // 2) Fold softmaxes + gamma into 32x32 mixing matrix
    weff_kernel<<<1, 64, 0, stream>>>(weights, gamma, weff);

    // 3) LayerNorm statistics per (batch, capsule) via weighted image sums
    stats_kernel<<<256, 256, 0, stream>>>(x, muB, rsB);

    // 4) Build P (8192 x 4608 bf16)
    build_p_kernel<<<M, 256, 0, stream>>>(x, weff, muB, rsB, P);

    // 5) GEMM1: a = sigmoid(bn1(P @ W1^T))  -> bf16 (8192 x 288)
    gemm_bf16_kernel<<<dim3(M / 128, N1 / 32), 256, 0, stream>>>(
        P, K0, nullptr, 0, W1t, N1,
        bn1S, bn1B, bn1M, bn1V, Abuf, nullptr, N1, 0);

    // 6) GEMM2: y = relu(bn2([P | a] @ W2^T)) -> f32 NCHW
    gemm_bf16_kernel<<<dim3(M / 128, N2 / 32), 256, 0, stream>>>(
        P, K0, Abuf, K1, W2t, N2,
        bn2S, bn2B, bn2M, bn2V, nullptr, out, N2, 1);
}
